// MultiOutputModel_70471823393248
// MI455X (gfx1250) — matrix operand order repaired and checked by reading, not compiled, not run on hardware
//
#include <hip/hip_runtime.h>
#include <hip/hip_bf16.h>

// MI455X (gfx1250) implementation.
//
// Roofline: ~2.7 GB of fp32 weights (eW1 2.15GB + eW2 0.54GB + backbone 44MB) must be
// streamed from HBM once per call -> memory bound, ~116us @ 23.3 TB/s. Total math is
// ~86 GFLOP at 32 FLOP per weight byte; that requires ~740 TFLOP/s to stay memory
// bound, which rules out fp32 V_WMMA_F32_16X16X4_F32 but is easy for
// V_WMMA_F32_16X16X32_BF16. Strategy: stream fp32 weights, v_cvt_pk_bf16_f32 them
// in-register, run bf16 WMMA with fp32 accumulation, keep all activations (<=1MB)
// in LDS/L2 as bf16.
//
// Backbone uses deterministic split-K (private fp32 partial buffers + ordered
// reduction, no atomics) to get >1000 waves in flight; Little's law needs ~18MB of
// outstanding loads to saturate 23.3 TB/s.

typedef __bf16 v16bf __attribute__((ext_vector_type(16)));
typedef float  v8f   __attribute__((ext_vector_type(8)));

#define B_DIM   64
#define E_DIM   4096
#define D1      2048
#define D2      1024
#define D3      512
#define H1      256
#define H2      128

// ---------- WMMA fragment loaders (layouts per CDNA5 ISA 7.12.2) ----------

// A: 16x32 bf16, row-major source. lanes0-15: M=lane, K = {0..7, 16..23};
// lanes16-31: M=lane-16, K = {8..15, 24..31}.
__device__ __forceinline__ v16bf load_a_f32(const float* __restrict__ S, int ld,
                                            int row, int k0, int half) {
  const float* p = S + (size_t)row * ld + k0 + 8 * half;
  v16bf a;
#pragma unroll
  for (int j = 0; j < 8; ++j) a[j] = (__bf16)p[j];
  p += 16;
#pragma unroll
  for (int j = 0; j < 8; ++j) a[8 + j] = (__bf16)p[j];
  return a;
}

__device__ __forceinline__ v16bf load_a_bf16(const __bf16* __restrict__ S, int ld,
                                             int row, int k0, int half) {
  const __bf16* p = S + (size_t)row * ld + k0 + 8 * half;
  v16bf a;
#pragma unroll
  for (int j = 0; j < 8; ++j) a[j] = p[j];
  p += 16;
#pragma unroll
  for (int j = 0; j < 8; ++j) a[8 + j] = p[j];
  return a;
}

// B: 32x16 bf16, row-major source W[ldb cols]. frag[j] = W[k0 + 8*half + j][col] for j<8,
// W[k0 + 16 + 8*half + j-8][col] for j>=8; col = lane%16 (same K order as A).
__device__ __forceinline__ v16bf load_b_f32(const float* __restrict__ W, int ldb,
                                            int k0, int col, int half) {
  const float* p = W + (size_t)(k0 + 8 * half) * ldb + col;
  v16bf b;
#pragma unroll
  for (int j = 0; j < 16; ++j) b[j] = (__bf16)p[(size_t)(j < 8 ? j : j + 8) * ldb];
  return b;
}

// ---------- backbone split-K partial GEMM ----------
// grid = (N/128, KSPLIT); 8 waves/block, one 16-col tile per wave, all 4 M-tiles of
// the batch per wave (weights read exactly once). Each K-split writes its own fp32
// partial [64,N] slice -> deterministic ordered reduction in the epilogue kernel.
template <bool ABF16>
__global__ __launch_bounds__(256) void backbone_partial(const void* __restrict__ Ain,
                                                        const float* __restrict__ W,
                                                        float* __restrict__ Pbuf,
                                                        int K, int N, int kps) {
  const int lane = threadIdx.x & 31;
  const int wave = threadIdx.x >> 5;
  const int half = lane >> 4;
  const int lp   = lane & 15;
  const int n0   = (blockIdx.x * 8 + wave) * 16;
  const int col  = n0 + lp;
  const int kb   = blockIdx.y * kps;
  const int ke   = kb + kps;

  v8f acc[4] = {};
#pragma unroll 2
  for (int k0 = kb; k0 < ke; k0 += 32) {
    v16bf b = load_b_f32(W, N, k0, col, half);
#pragma unroll
    for (int mt = 0; mt < 4; ++mt) {
      v16bf a;
      if constexpr (ABF16)
        a = load_a_bf16((const __bf16*)Ain, K, mt * 16 + lp, k0, half);
      else
        a = load_a_f32((const float*)Ain, K, mt * 16 + lp, k0, half);
      acc[mt] = __builtin_amdgcn_wmma_f32_16x16x32_bf16(
          false, a, false, b, (short)0, acc[mt], false, false);
    }
  }

  float* P = Pbuf + (size_t)blockIdx.y * B_DIM * N;
#pragma unroll
  for (int mt = 0; mt < 4; ++mt) {
#pragma unroll
    for (int i = 0; i < 8; ++i) {
      const int row = mt * 16 + i + 8 * half;   // C/D layout: VGPR i -> M = i + 8*half
      P[(size_t)row * N + col] = acc[mt][i];
    }
  }
}

// ---------- backbone epilogue: ordered split-K reduce + bias + ReLU + bf16 ----------
__global__ __launch_bounds__(256) void bias_relu_cvt(const float* __restrict__ Pbuf,
                                                     const float* __restrict__ bias,
                                                     __bf16* __restrict__ out,
                                                     int N, int ksplit) {
  const int idx = blockIdx.x * 256 + threadIdx.x;   // grid sized to exactly 64*N
  float s = 0.f;
  for (int p = 0; p < ksplit; ++p)                  // fixed order -> deterministic
    s += Pbuf[(size_t)p * B_DIM * N + idx];
  const int col = idx & (N - 1);                    // N is a power of two
  float v = s + bias[col];
  out[idx] = (__bf16)(v > 0.f ? v : 0.f);
}

// ---------- expert kernel: one workgroup per expert ----------
// h3[64,512] bf16 staged in LDS; layer1 (512->256) and layer2 (256->128) via WMMA
// with bf16 activations in LDS; layer3 (128->1) via VALU dot products.
__global__ __launch_bounds__(256) void expert_kernel(const __bf16* __restrict__ h3g,
                                                     const float* __restrict__ eW1,
                                                     const float* __restrict__ eb1,
                                                     const float* __restrict__ eW2,
                                                     const float* __restrict__ eb2,
                                                     const float* __restrict__ eW3,
                                                     const float* __restrict__ eb3,
                                                     float* __restrict__ out) {
  __shared__ __bf16 sH[B_DIM * D3];    // 64 KiB
  __shared__ __bf16 sA1[B_DIM * H1];   // 32 KiB
  __shared__ __bf16 sA2[B_DIM * H2];   // 16 KiB

  const int e    = blockIdx.x;
  const int tid  = threadIdx.x;
  const int lane = tid & 31;
  const int wave = tid >> 5;
  const int half = lane >> 4;
  const int lp   = lane & 15;

  // stage h3 (bf16) into LDS as dwords
  {
    const uint32_t* src = (const uint32_t*)h3g;
    uint32_t* dst = (uint32_t*)sH;
    for (int i = tid; i < B_DIM * D3 / 2; i += 256) dst[i] = src[i];
  }
  __syncthreads();

  // ---- layer 1: a1[64,256] = relu(h3 @ eW1[e] + eb1[e]) ; 16 n-tiles, 2 per wave
  const float* W1 = eW1 + (size_t)e * D3 * H1;
#pragma unroll
  for (int rep = 0; rep < 2; ++rep) {
    const int n0  = (wave + rep * 8) * 16;
    const int col = n0 + lp;
    v8f acc[4] = {};
#pragma unroll 2
    for (int k0 = 0; k0 < D3; k0 += 32) {
      v16bf b = load_b_f32(W1, H1, k0, col, half);
#pragma unroll
      for (int mt = 0; mt < 4; ++mt) {
        v16bf a = load_a_bf16(sH, D3, mt * 16 + lp, k0, half);
        acc[mt] = __builtin_amdgcn_wmma_f32_16x16x32_bf16(
            false, a, false, b, (short)0, acc[mt], false, false);
      }
    }
    const float bv = eb1[(size_t)e * H1 + col];
#pragma unroll
    for (int mt = 0; mt < 4; ++mt) {
#pragma unroll
      for (int i = 0; i < 8; ++i) {
        const int row = mt * 16 + i + 8 * half;
        float v = acc[mt][i] + bv;
        v = v > 0.f ? v : 0.f;
        sA1[row * H1 + col] = (__bf16)v;
      }
    }
  }
  __syncthreads();

  // ---- layer 2: a2[64,128] = relu(a1 @ eW2[e] + eb2[e]) ; 8 n-tiles, 1 per wave
  {
    const float* W2 = eW2 + (size_t)e * H1 * H2;
    const int n0  = wave * 16;
    const int col = n0 + lp;
    v8f acc[4] = {};
#pragma unroll 2
    for (int k0 = 0; k0 < H1; k0 += 32) {
      v16bf b = load_b_f32(W2, H2, k0, col, half);
#pragma unroll
      for (int mt = 0; mt < 4; ++mt) {
        v16bf a = load_a_bf16(sA1, H1, mt * 16 + lp, k0, half);
        acc[mt] = __builtin_amdgcn_wmma_f32_16x16x32_bf16(
            false, a, false, b, (short)0, acc[mt], false, false);
      }
    }
    const float bv = eb2[(size_t)e * H2 + col];
#pragma unroll
    for (int mt = 0; mt < 4; ++mt) {
#pragma unroll
      for (int i = 0; i < 8; ++i) {
        const int row = mt * 16 + i + 8 * half;
        float v = acc[mt][i] + bv;
        v = v > 0.f ? v : 0.f;
        sA2[row * H2 + col] = (__bf16)v;
      }
    }
  }
  __syncthreads();

  // ---- layer 3: out[b,e] = a2[b,:] . eW3[e,:] + eb3[e]
  if (tid < B_DIM) {
    const float* w3 = eW3 + (size_t)e * H2;
    float s = 0.f;
#pragma unroll 8
    for (int k = 0; k < H2; ++k) s += (float)sA2[tid * H2 + k] * w3[k];
    out[(size_t)tid * E_DIM + e] = s + eb3[e];
  }
}

// ---------- host launcher ----------
extern "C" void kernel_launch(void* const* d_in, const int* in_sizes, int n_in,
                              void* d_out, int out_size, void* d_ws, size_t ws_size,
                              hipStream_t stream) {
  (void)in_sizes; (void)n_in; (void)out_size; (void)ws_size;

  const float* x    = (const float*)d_in[0];
  const float* fcW1 = (const float*)d_in[1];
  const float* fcb1 = (const float*)d_in[2];
  const float* fcW2 = (const float*)d_in[3];
  const float* fcb2 = (const float*)d_in[4];
  const float* fcW3 = (const float*)d_in[5];
  const float* fcb3 = (const float*)d_in[6];
  const float* eW1  = (const float*)d_in[7];
  const float* eb1  = (const float*)d_in[8];
  const float* eW2  = (const float*)d_in[9];
  const float* eb2  = (const float*)d_in[10];
  const float* eW3  = (const float*)d_in[11];
  const float* eb3  = (const float*)d_in[12];
  float* out = (float*)d_out;

  // workspace: bf16 h1[64,2048], h2[64,1024], h3[64,512] (448 KiB) then fp32
  // split-K partial buffer (max 8 * 64*2048 f32 = 4 MiB).
  __bf16* h1 = (__bf16*)d_ws;
  __bf16* h2 = h1 + (size_t)B_DIM * D1;
  __bf16* h3 = h2 + (size_t)B_DIM * D2;
  float*  P  = (float*)(h3 + (size_t)B_DIM * D3);

  // ---- backbone with deterministic split-K ----
  // fc1: [64,4096] @ [4096,2048], KSPLIT=8 -> 128 blocks, 1024 waves
  backbone_partial<false><<<dim3(D1 / 128, 8), 256, 0, stream>>>(x, fcW1, P, 4096, D1, 4096 / 8);
  bias_relu_cvt<<<(B_DIM * D1) / 256, 256, 0, stream>>>(P, fcb1, h1, D1, 8);
  // fc2: [64,2048] @ [2048,1024], KSPLIT=4
  backbone_partial<true><<<dim3(D2 / 128, 4), 256, 0, stream>>>(h1, fcW2, P, D1, D2, D1 / 4);
  bias_relu_cvt<<<(B_DIM * D2) / 256, 256, 0, stream>>>(P, fcb2, h2, D2, 4);
  // fc3: [64,1024] @ [1024,512], KSPLIT=4
  backbone_partial<true><<<dim3(D3 / 128, 4), 256, 0, stream>>>(h2, fcW3, P, D2, D3, D2 / 4);
  bias_relu_cvt<<<(B_DIM * D3) / 256, 256, 0, stream>>>(P, fcb3, h3, D3, 4);

  // ---- experts: one workgroup per expert head ----
  expert_kernel<<<E_DIM, 256, 0, stream>>>(h3, eW1, eb1, eW2, eb2, eW3, eb3, out);
}
